// GatedDeltaNet_7894149890710
// MI455X (gfx1250) — matrix operand order repaired and checked by reading, not compiled, not run on hardware
//
#include <hip/hip_runtime.h>
#include <hip/hip_bf16.h>

// ---------------- problem constants ----------------
constexpr int Bc  = 2;
constexpr int Sc  = 2048;
constexpr int Hc  = 2048;
constexpr int DKc = 128;
constexpr int DVc = 128;
constexpr int HKc = 16;
constexpr int HVc = 32;
constexpr int Gc  = 2;         // HV / HK
constexpr int Kw  = 4;         // conv kernel width
constexpr int FQKVZ = 2 * HKc * DKc + 2 * HVc * DVc;   // 12288
constexpr int FBA   = 2 * HVc;                          // 64
constexpr int Cch   = 2 * HKc * DKc + HVc * DVc;        // 8192 conv channels
constexpr int BS    = Bc * Sc;                          // 4096 rows
constexpr float EPSf = 1e-6f;

// ---------------- vector types ----------------
typedef __attribute__((ext_vector_type(16))) __bf16 v16bf;
typedef __attribute__((ext_vector_type(8)))  __bf16 v8bf;
typedef __attribute__((ext_vector_type(4)))  __bf16 v4bf;
typedef __attribute__((ext_vector_type(8)))  float  v8f;
typedef __attribute__((ext_vector_type(4)))  float  v4f;
typedef int v4i_gcc __attribute__((vector_size(16)));   // matches builtin proto

__device__ __forceinline__ __bf16 f2bf(float f) {
    unsigned u = __builtin_bit_cast(unsigned, f);
    u += 0x7FFFu + ((u >> 16) & 1u);            // round-to-nearest-even
    unsigned short h = (unsigned short)(u >> 16);
    return __builtin_bit_cast(__bf16, h);
}

// ---------------- async-to-LDS staging (CDNA5 path w/ fallback) --------
#if defined(__HIP_DEVICE_COMPILE__) && \
    __has_builtin(__builtin_amdgcn_global_load_async_to_lds_b128) && \
    __has_builtin(__builtin_amdgcn_s_wait_asynccnt)
#define USE_ASYNC_LDS 1
#else
#define USE_ASYNC_LDS 0
#endif

__device__ __forceinline__ void stage16(__bf16* lds_dst, const __bf16* g_src) {
#if USE_ASYNC_LDS
    __builtin_amdgcn_global_load_async_to_lds_b128(
        (__attribute__((address_space(1))) v4i_gcc*)(v4i_gcc*)g_src,
        (__attribute__((address_space(3))) v4i_gcc*)(v4i_gcc*)lds_dst,
        0, 0);
#else
    *(v8bf*)lds_dst = *(const v8bf*)g_src;      // global_load_b128 + ds_store_b128
#endif
}

__device__ __forceinline__ void stage_wait() {
#if USE_ASYNC_LDS
    __builtin_amdgcn_s_wait_asynccnt(0);
#endif
    __syncthreads();
}

// ---------------- bf16 pack kernels ----------------
// elementwise f32 -> bf16 (4 per thread)
__global__ __launch_bounds__(256)
void pack_bf16(const float* __restrict__ in, __bf16* __restrict__ out) {
    size_t i4 = ((size_t)blockIdx.x * 256 + threadIdx.x) * 4;
    v4f v = *(const v4f*)(in + i4);
    v4bf o = { f2bf(v.x), f2bf(v.y), f2bf(v.z), f2bf(v.w) };
    *(v4bf*)(out + i4) = o;
}

// transpose + convert: in = [Kd x N] f32 row-major -> out = [N x Kd] bf16
__global__ __launch_bounds__(256)
void pack_bf16_T(const float* __restrict__ in, __bf16* __restrict__ out,
                 int Kd, int N) {
    size_t idx = (size_t)blockIdx.x * 256 + threadIdx.x;   // over N*Kd
    int k = (int)(idx % Kd);
    size_t n = idx / Kd;
    out[idx] = f2bf(in[(size_t)k * N + n]);
}

// =====================================================================
// GEMM: C[M,N](f32) = A[M,K](bf16, row-major) * B[K,N] where B is given
// pre-transposed as Bt[N,K](bf16, row-major).
// Block = 256 thr = 8 waves over a 64M x 64N tile, BK = 32.
// Wave (wm,wn): wm in {0,1} -> 32 M-rows, wn in {0..3} -> 16 N-cols.
// 2 x v_wmma_f32_16x16x32_bf16 per wave per K-step, fed from LDS with
// ds_load_b128 fragment reads matching the ISA bf16 A/B lane layouts.
// Requires M%64==0, N%64==0, K%32==0 (holds for all three GEMMs).
// =====================================================================
constexpr int BM = 64, BN = 64, BK = 32;

__device__ __forceinline__ v16bf ldfrag(const __bf16* p0, const __bf16* p1) {
    v8bf lo = *(const v8bf*)p0;
    v8bf hi = *(const v8bf*)p1;
    return __builtin_shufflevector(lo, hi, 0,1,2,3,4,5,6,7,8,9,10,11,12,13,14,15);
}

__global__ __launch_bounds__(256)
void gemm_bf16_wmma(const __bf16* __restrict__ Abf,
                    const __bf16* __restrict__ Btb,
                    float* __restrict__ Cm, int N, int Kd) {
    __shared__ __attribute__((aligned(16))) __bf16 As[BM * BK];
    __shared__ __attribute__((aligned(16))) __bf16 Bs[BN * BK];

    const int tid  = threadIdx.x;
    const int lane = tid & 31;
    const int w    = tid >> 5;
    const int wm   = w & 1, wn = w >> 1;
    const int lm   = lane & 15, lh = lane >> 4;
    const int mBlock = blockIdx.x * BM, nBlock = blockIdx.y * BN;

    // staging: each thread moves one 16B chunk of A-tile and of B-tile
    const int srow = tid >> 2;          // 0..63
    const int scol = (tid & 3) * 8;     // 0,8,16,24 (bf16 elements)
    const __bf16* Ag = Abf + (size_t)(mBlock + srow) * Kd + scol;
    const __bf16* Bg = Btb + (size_t)(nBlock + srow) * Kd + scol;
    __bf16* Al = &As[srow * BK + scol];
    __bf16* Bl = &Bs[srow * BK + scol];

    v8f acc0 = {}, acc1 = {};
    for (int k0 = 0; k0 < Kd; k0 += BK) {
        stage16(Al, Ag + k0);
        stage16(Bl, Bg + k0);
        __builtin_prefetch(Ag + k0 + BK, 0, 3);   // global_prefetch_b8
        stage_wait();

        // A fragments (16-bit A 16x32): lane L: M=L%16,
        //   e<8 -> K = lh*8 + e ; e>=8 -> K = 16 + lh*8 + (e-8)
        const int ar0 = (wm * 32 + lm) * BK;
        const int ar1 = (wm * 32 + 16 + lm) * BK;
        v16bf a0 = ldfrag(&As[ar0 + lh * 8], &As[ar0 + 16 + lh * 8]);
        v16bf a1 = ldfrag(&As[ar1 + lh * 8], &As[ar1 + 16 + lh * 8]);
        // B fragment (16-bit B 32x16): lane L: N=L%16,
        //   e<8 -> K = lh*8 + e ; e>=8 -> K = 16 + lh*8 + (e-8)
        const int br = (wn * 16 + lm) * BK + lh * 8;
        v16bf b = ldfrag(&Bs[br], &Bs[br + 16]);

        acc0 = __builtin_amdgcn_wmma_f32_16x16x32_bf16(
            false, a0, false, b, (short)0, acc0, false, false);
        acc1 = __builtin_amdgcn_wmma_f32_16x16x32_bf16(
            false, a1, false, b, (short)0, acc1, false, false);
        __syncthreads();
    }

    // C/D layout: VGPR r, lane L -> M = (L>>4)*8 + r, N = L%16
    const int mB = mBlock + wm * 32, nB = nBlock + wn * 16;
#pragma unroll
    for (int r = 0; r < 8; ++r) {
        Cm[(size_t)(mB + lh * 8 + r) * N + nB + lm]      = acc0[r];
        Cm[(size_t)(mB + 16 + lh * 8 + r) * N + nB + lm] = acc1[r];
    }
}

// =====================================================================
// Depthwise causal conv1d (width 4) + bias + SiLU.
// =====================================================================
__device__ __forceinline__ int qkvz_col_of_channel(int c) {
    if (c < HKc * DKc) {                       // q
        return (c >> 7) * 768 + (c & 127);
    } else if (c < 2 * HKc * DKc) {            // k
        int cc = c - HKc * DKc;
        return (cc >> 7) * 768 + 128 + (cc & 127);
    } else {                                   // v
        int cc = c - 2 * HKc * DKc;
        int hv = cc >> 7, d = cc & 127;
        return (hv >> 1) * 768 + 256 + (hv & 1) * 128 + d;
    }
}

__global__ __launch_bounds__(256)
void conv_silu_kernel(const float* __restrict__ qkvz,
                      const float* __restrict__ convw,
                      const float* __restrict__ convb,
                      float* __restrict__ mixed) {
    size_t idx = (size_t)blockIdx.x * 256 + threadIdx.x;   // B*S*C threads
    int c  = (int)(idx % Cch);
    size_t bs = idx / Cch;
    int s  = (int)(bs % Sc);
    int b  = (int)(bs / Sc);
    int col = qkvz_col_of_channel(c);

    float acc = convb[c];
#pragma unroll
    for (int j = 0; j < Kw; ++j) {
        int sp = s - (Kw - 1) + j;
        if (sp >= 0) {
            float xin = qkvz[((size_t)b * Sc + sp) * FQKVZ + col];
            acc += xin * convw[c * Kw + j];
        }
    }
    float y = acc / (1.f + __expf(-acc));      // SiLU
    mixed[((size_t)b * Sc + s) * Cch + c] = y;
}

// =====================================================================
// Gating: g = -exp(a_log)*softplus(a+dt_bias), beta = sigmoid(b)
// =====================================================================
__global__ __launch_bounds__(256)
void gate_kernel(const float* __restrict__ ba,
                 const float* __restrict__ a_log,
                 const float* __restrict__ dt_bias,
                 float* __restrict__ gbuf, float* __restrict__ bbuf) {
    int idx = blockIdx.x * 256 + threadIdx.x;  // B*S*HV
    int hv = idx & (HVc - 1);
    int bs = idx >> 5;
    int hk = hv >> 1, j = hv & 1;
    const float* row = ba + (size_t)bs * FBA + hk * (2 * Gc);
    float bv = row[j];
    float av = row[Gc + j] + dt_bias[hv];
    float sp = (av > 20.f) ? av : log1pf(__expf(av));
    gbuf[(size_t)bs * HVc + hv] = -__expf(a_log[hv]) * sp;
    bbuf[(size_t)bs * HVc + hv] = 1.f / (1.f + __expf(-bv));
}

// =====================================================================
// L2-normalize q (then * DK^-0.5) and k rows in-place in mixed buffer.
// =====================================================================
__global__ __launch_bounds__(128)
void qknorm_kernel(float* __restrict__ mixed) {
    __shared__ float rq[128], rk[128];
    int blk = blockIdx.x;                      // B*S*HK blocks
    int hk  = blk % HKc;
    size_t bs = blk / HKc;
    float* qp = mixed + bs * Cch + hk * 128;
    float* kp = qp + HKc * DKc;
    int t = threadIdx.x;
    float q = qp[t], k = kp[t];
    rq[t] = q * q; rk[t] = k * k;
    __syncthreads();
    for (int off = 64; off > 0; off >>= 1) {
        if (t < off) { rq[t] += rq[t + off]; rk[t] += rk[t + off]; }
        __syncthreads();
    }
    qp[t] = q * rsqrtf(rq[0] + EPSf) * 0.08838834764831845f;  // DK^-0.5
    kp[t] = k * rsqrtf(rk[0] + EPSf);
}

// =====================================================================
// Sequential delta-rule scan. One block per (b,hv): 128x128 f32 state in
// LDS (64 KB of 320 KB/WGP), column t owned by lane t. Fused epilogue:
// y = o * silu(z), RMSNorm over DV, * norm_w.
// =====================================================================
__global__ __launch_bounds__(128)
void scan_kernel(const float* __restrict__ mixed,   // normalized q/k + v
                 const float* __restrict__ qkvz,    // for z
                 const float* __restrict__ gbuf,
                 const float* __restrict__ bbuf,
                 const float* __restrict__ norm_w,
                 float* __restrict__ ybuf) {
    extern __shared__ float smem[];
    float* St  = smem;                 // 128*128
    float* qsh = smem + 128 * 128;     // 128
    float* ksh = qsh + 128;
    float* vsh = ksh + 128;
    float* red = vsh + 128;            // 128
    __shared__ float gb[2];

    const int t   = threadIdx.x;
    const int blk = blockIdx.x;        // B*HV = 64
    const int b   = blk >> 5;
    const int hv  = blk & 31;
    const int hk  = hv >> 1, j = hv & 1;
    const float nw = norm_w[t];

    for (int r = 0; r < 128; ++r) St[r * 128 + t] = 0.f;
    __syncthreads();

    for (int s = 0; s < Sc; ++s) {
        size_t base = (size_t)b * Sc + s;
        const float* mrow = mixed + base * Cch;
        qsh[t] = mrow[hk * 128 + t];
        ksh[t] = mrow[HKc * DKc + hk * 128 + t];
        vsh[t] = mrow[2 * HKc * DKc + hv * 128 + t];
        if (t == 0) {
            gb[0] = gbuf[base * HVc + hv];
            gb[1] = bbuf[base * HVc + hv];
        }
        __syncthreads();

        const float eg   = __expf(gb[0]);
        const float beta = gb[1];

        float acc = 0.f;
#pragma unroll 8
        for (int kk = 0; kk < 128; ++kk) acc += ksh[kk] * St[kk * 128 + t];
        const float dv = beta * (vsh[t] - eg * acc);

        float o = 0.f;
#pragma unroll 8
        for (int kk = 0; kk < 128; ++kk) {
            float sn = eg * St[kk * 128 + t] + ksh[kk] * dv;
            St[kk * 128 + t] = sn;
            o += qsh[kk] * sn;
        }

        float z = qkvz[base * FQKVZ + hk * 768 + 512 + j * 128 + t];
        float y = o * (z / (1.f + __expf(-z)));
        red[t] = y * y;
        __syncthreads();
        for (int off = 64; off > 0; off >>= 1) {
            if (t < off) red[t] += red[t + off];
            __syncthreads();
        }
        float var = red[0] * (1.f / 128.f);
        y = y * rsqrtf(var + EPSf) * nw;
        ybuf[(base * HVc + hv) * 128 + t] = y;
        __syncthreads();
    }
}

// =====================================================================
// Launcher
// =====================================================================
extern "C" void kernel_launch(void* const* d_in, const int* in_sizes, int n_in,
                              void* d_out, int out_size, void* d_ws, size_t ws_size,
                              hipStream_t stream) {
    const float* x       = (const float*)d_in[0];
    const float* w_qkvz  = (const float*)d_in[1];
    const float* w_ba    = (const float*)d_in[2];
    const float* conv_w  = (const float*)d_in[3];
    const float* conv_b  = (const float*)d_in[4];
    const float* a_log   = (const float*)d_in[5];
    const float* dt_bias = (const float*)d_in[6];
    const float* norm_w  = (const float*)d_in[7];
    const float* w_o     = (const float*)d_in[8];
    float* out = (float*)d_out;

    char* ws = (char*)d_ws;
    float*  qkvz  = (float*)ws;  ws += (size_t)BS * FQKVZ * sizeof(float);   // 201 MB
    float*  ba    = (float*)ws;  ws += (size_t)BS * FBA   * sizeof(float);
    float*  mixed = (float*)ws;  ws += (size_t)BS * Cch   * sizeof(float);   // 134 MB
    float*  gbuf  = (float*)ws;  ws += (size_t)BS * HVc   * sizeof(float);
    float*  bbuf  = (float*)ws;  ws += (size_t)BS * HVc   * sizeof(float);
    float*  ybuf  = (float*)ws;  ws += (size_t)BS * HVc * DVc * sizeof(float); // 67 MB
    __bf16* xbf   = (__bf16*)ws; ws += (size_t)BS * Hc        * sizeof(__bf16);
    __bf16* wqT   = (__bf16*)ws; ws += (size_t)FQKVZ * Hc     * sizeof(__bf16); // 50 MB
    __bf16* wbT   = (__bf16*)ws; ws += (size_t)FBA * Hc       * sizeof(__bf16);
    __bf16* woT   = (__bf16*)ws; ws += (size_t)Hc * (HVc*DVc) * sizeof(__bf16); // 17 MB
    __bf16* ybf   = (__bf16*)ws; ws += (size_t)BS * HVc * DVc * sizeof(__bf16);

    dim3 blk256(256);

    // ---- pack operands to bf16 (A row-major, B transposed to NxK) ----
    pack_bf16 <<<(size_t)BS * Hc / 1024, blk256, 0, stream>>>(x, xbf);
    pack_bf16_T<<<(size_t)FQKVZ * Hc / 256, blk256, 0, stream>>>(w_qkvz, wqT, Hc, FQKVZ);
    pack_bf16_T<<<(size_t)FBA * Hc / 256, blk256, 0, stream>>>(w_ba, wbT, Hc, FBA);
    pack_bf16_T<<<(size_t)Hc * (HVc*DVc) / 256, blk256, 0, stream>>>(w_o, woT, HVc*DVc, Hc);

    // 1) qkvz = x @ w_qkvz   (4096 x 2048 x 12288)
    gemm_bf16_wmma<<<dim3(BS / BM, FQKVZ / BN), blk256, 0, stream>>>(
        xbf, wqT, qkvz, FQKVZ, Hc);
    // 2) ba = x @ w_ba       (4096 x 2048 x 64)
    gemm_bf16_wmma<<<dim3(BS / BM, FBA / BN), blk256, 0, stream>>>(
        xbf, wbT, ba, FBA, Hc);
    // 3) depthwise causal conv + SiLU
    conv_silu_kernel<<<(size_t)BS * Cch / 256, blk256, 0, stream>>>(
        qkvz, conv_w, conv_b, mixed);
    // 4) gates
    gate_kernel<<<BS * HVc / 256, blk256, 0, stream>>>(
        ba, a_log, dt_bias, gbuf, bbuf);
    // 5) q/k L2 norm (in place)
    qknorm_kernel<<<BS * HKc, dim3(128), 0, stream>>>(mixed);
    // 6) sequential delta-rule scan with fused gate + RMSNorm
    size_t scan_smem = (size_t)(128 * 128 + 4 * 128) * sizeof(float); // ~66 KB
    scan_kernel<<<Bc * HVc, dim3(128), scan_smem, stream>>>(
        mixed, qkvz, gbuf, bbuf, norm_w, ybuf);
    // 7) out = y @ w_o       (4096 x 4096 x 2048)
    pack_bf16<<<(size_t)BS * HVc * DVc / 1024, blk256, 0, stream>>>(ybuf, ybf);
    gemm_bf16_wmma<<<dim3(BS / BM, Hc / BN), blk256, 0, stream>>>(
        ybf, woT, out, Hc, HVc * DVc);
}
